// GridTorch_53257594470618
// MI455X (gfx1250) — compile-verified, matrix operand order checked by reading, not run on hardware
//
#include <hip/hip_runtime.h>
#include <hip/hip_bf16.h>
#include <stdint.h>

// ---------------------------------------------------------------------------
// GridTorch LSTM on MI455X (gfx1250, wave32, WMMA bf16 16x16x32)
//
// B=1024 -> 64 workgroups x 16 rows; each WG runs all 100 steps (recurrence is
// independent per batch row).  Combined [x_pad(32); h(128)] x [Wpad; U]
// (160x512) gate GEMM.  Weights repacked to bf16 WMMA B-fragment order; U+Wbn
// staged into LDS via global_load_async_to_lds_b128 (ASYNCcnt), W_pc/W_hd
// stream from L2.  Cell state c lives in accumulator-layout VGPRs for all 100
// steps.  3 split barriers per step with VALU / global-store work hidden
// between signal and wait.  GEMM fragment loads software-pipelined; with
// amdgpu_waves_per_eu(2,2) (exactly 1 WG/WGP due to 243KB LDS) the scheduler
// has ~half the VGPR file per wave to keep the pipeline in registers.
// All 320MB of outputs stored non-temporal.
// ---------------------------------------------------------------------------

typedef __attribute__((ext_vector_type(16))) __bf16 v16bf;
typedef __attribute__((ext_vector_type(8)))  __bf16 v8bf;
typedef __attribute__((ext_vector_type(8)))  float  v8f;

#define T_STEPS 100
#define BATCH   1024
#define HS      128
#define NBN     256
#define NPC     256
#define NHD     12

// packed bf16 weight regions (element offsets)
#define PK_WU    0          // [160x512] K-tiles=5,  N-tiles=32 -> 81920
#define PK_WBN   81920      // [128x256] K-tiles=4,  N-tiles=16 -> 32768
#define PK_WPC   114688     // [256x256] K-tiles=8,  N-tiles=16 -> 65536
#define PK_WHD   180224     // [256x16]  K-tiles=8,  N-tiles=1  -> 4096
#define PK_TOTAL 184320

#define WS_H0_BYTES  (PK_TOTAL * 2)                    // 368640
#define WS_C0_BYTES  (WS_H0_BYTES + BATCH * HS * 4)    // 892928

// LDS layout (bytes)
#define L_WU     0
#define L_WBN    163840                 // 81920 bf16
#define L_ABUF   229376                 // [16][SA] bf16 (x-pad 0..31, h 32..159)
#define SA       168                    // padded row stride (bank-spread)
#define L_BNBUF  (L_ABUF + 16 * SA * 2) // 234752, [16][SB] bf16
#define SB       264
#define L_TOTAL  (L_BNBUF + 16 * SB * 2) // 243200 bytes dynamic LDS

// ---------------------------------------------------------------------------

__device__ __forceinline__ __bf16 f2bf(float f) {
  union { float f; uint32_t u; } v; v.f = f;
  uint32_t r = (v.u + 0x7FFFu + ((v.u >> 16) & 1u)) >> 16;
  uint16_t s = (uint16_t)r;
  __bf16 b; __builtin_memcpy(&b, &s, 2); return b;
}

__device__ __forceinline__ v16bf cat16(v8bf a, v8bf b) {
  return __builtin_shufflevector(a, b, 0,1,2,3,4,5,6,7,8,9,10,11,12,13,14,15);
}

// A-fragment (16x32 bf16): lane m=l%16; K order {kb..kb+7, 16+kb..16+kb+7},
// kb = 8*(l/16).  base is row-major bf16 with row stride `stride` elements.
__device__ __forceinline__ v16bf ldA(const __bf16* base, int stride, int lane, int kbase) {
  int m  = lane & 15;
  int hi = (lane >> 4) << 3;
  const __bf16* p = base + m * stride + kbase + hi;
  return cat16(*(const v8bf*)p, *(const v8bf*)(p + 16));
}

// B-fragment: packed lane-major, 16 contiguous bf16 per lane per tile.
__device__ __forceinline__ v16bf ldB(const __bf16* tile, int lane) {
  const __bf16* p = tile + (lane << 4);
  return cat16(*(const v8bf*)p, *(const v8bf*)(p + 8));
}

__device__ __forceinline__ v8f wmma_bf(v16bf a, v16bf b, v8f c) {
  return __builtin_amdgcn_wmma_f32_16x16x32_bf16(false, a, false, b, (short)0, c,
                                                 false, false);
}

__device__ __forceinline__ float sigm(float x) { return 1.0f / (1.0f + __expf(-x)); }
__device__ __forceinline__ float tanh_(float x) {
  float e = __expf(-2.0f * x);
  return (1.0f - e) / (1.0f + e);
}

// --- gfx1250 sync / async helpers ------------------------------------------

__device__ __forceinline__ void bar_signal() {
#if __has_builtin(__builtin_amdgcn_s_barrier_signal)
  __builtin_amdgcn_s_barrier_signal(-1);
#else
  asm volatile("s_barrier_signal -1" ::: "memory");
#endif
}
__device__ __forceinline__ void bar_wait() {
#if __has_builtin(__builtin_amdgcn_s_barrier_wait)
  __builtin_amdgcn_s_barrier_wait(-1);
#else
  asm volatile("s_barrier_wait -1" ::: "memory");
#endif
}
__device__ __forceinline__ void wait_dscnt0() {
  asm volatile("s_wait_dscnt 0x0" ::: "memory");
}
__device__ __forceinline__ void wait_asynccnt0() {
#if __has_builtin(__builtin_amdgcn_s_wait_asynccnt)
  __builtin_amdgcn_s_wait_asynccnt(0);
#else
  asm volatile("s_wait_asynccnt 0x0" ::: "memory");
#endif
}
// One 16B async global->LDS DMA per lane (ASYNCcnt tracked, no VGPR bounce).
__device__ __forceinline__ void async_g2l_b128(uint32_t lds_off, uint64_t gbase,
                                               uint32_t voff) {
  asm volatile("global_load_async_to_lds_b128 %0, %1, %2"
               :: "v"(lds_off), "v"(voff), "s"(gbase) : "memory");
}

// (packed element index, N-tile count) -> (k, n) of source matrix.
__device__ __forceinline__ void unpack_pk(int p, int Nt, int& k, int& n) {
  int tile = p >> 9;
  int rem  = p & 511;
  int lane = rem >> 4;
  int e    = rem & 15;
  int kt = tile / Nt, nt = tile - kt * Nt;
  n = nt * 16 + (lane & 15);
  int kb = (lane < 16) ? 0 : 8;
  k = kt * 32 + ((e < 8) ? (e + kb) : (16 + (e - 8) + kb));
}

// ---------------------------------------------------------------------------
// Kernel 1: repack all weights to bf16 WMMA-fragment layout in workspace.
__global__ void __launch_bounds__(256) k_pack(
    const float* __restrict__ W, const float* __restrict__ U,
    const float* __restrict__ Wbn, const float* __restrict__ Wpc,
    const float* __restrict__ Whd, __bf16* __restrict__ out) {
  int p = blockIdx.x * 256 + threadIdx.x;
  if (p >= PK_TOTAL) return;
  int k, n; float v;
  if (p < PK_WBN) {                      // combined [Wpad(32x512); U(128x512)]
    unpack_pk(p, 32, k, n);
    v = (k < 32) ? ((k < 3) ? W[k * 512 + n] : 0.0f) : U[(k - 32) * 512 + n];
  } else if (p < PK_WPC) {
    unpack_pk(p - PK_WBN, 16, k, n);
    v = Wbn[k * 256 + n];
  } else if (p < PK_WHD) {
    unpack_pk(p - PK_WPC, 16, k, n);
    v = Wpc[k * 256 + n];
  } else {
    unpack_pk(p - PK_WHD, 1, k, n);
    v = (n < NHD) ? Whd[k * NHD + n] : 0.0f;
  }
  out[p] = f2bf(v);
}

// Kernel 2: h0 = init_conds @ W_ih + b_ih ; c0 likewise (one-time small GEMM).
__global__ void __launch_bounds__(256) k_init(
    const float* __restrict__ ic, const float* __restrict__ Wih,
    const float* __restrict__ bih, const float* __restrict__ Wic,
    const float* __restrict__ bic, float* __restrict__ h0,
    float* __restrict__ c0) {
  int idx = blockIdx.x * 256 + threadIdx.x;   // 0 .. 1024*128-1
  int b = idx >> 7, n = idx & 127;
  const float* row = ic + b * 268;
  float ah = bih[n], ac = bic[n];
  for (int k = 0; k < 268; ++k) {
    float v = row[k];
    ah = fmaf(v, Wih[k * 128 + n], ah);
    ac = fmaf(v, Wic[k * 128 + n], ac);
  }
  h0[idx] = ah;
  c0[idx] = ac;
}

// Kernel 3: persistent per-B-tile LSTM, all 100 steps.
// 1 WG/WGP (LDS-bound) => exactly 2 waves/SIMD; tell the register allocator.
__global__ void __launch_bounds__(256)
__attribute__((amdgpu_waves_per_eu(2, 2)))
k_lstm(
    const float* __restrict__ x, const float* __restrict__ bvec,
    const float* __restrict__ bpc, const float* __restrict__ bhd,
    const __bf16* __restrict__ pk, const float* __restrict__ h0,
    const float* __restrict__ c0, float* __restrict__ oHd,
    float* __restrict__ oPc, float* __restrict__ oBn,
    float* __restrict__ oH, float* __restrict__ oC) {
  extern __shared__ char smem[];
  __bf16* wuL  = (__bf16*)(smem + L_WU);
  __bf16* wbnL = (__bf16*)(smem + L_WBN);
  __bf16* Ab   = (__bf16*)(smem + L_ABUF);
  __bf16* BnB  = (__bf16*)(smem + L_BNBUF);

  const int tid  = threadIdx.x;
  const int lane = tid & 31;
  const int w    = tid >> 5;           // wave 0..7
  const int bt   = blockIdx.x;         // B-tile 0..63
  const int N    = lane & 15;          // C/D column within 16-wide tile
  const int mhi  = (lane >> 4) << 3;   // C/D row offset 0 or 8

  // Stage WU + Wbn into LDS via async LDS-DMA: 229376 B = 14336 x 16B.
  {
    uint32_t lbase = (uint32_t)(uintptr_t)smem;
    uint64_t gbase = (uint64_t)(uintptr_t)pk;
    for (int i = tid; i < 14336; i += 256)
      async_g2l_b128(lbase + (uint32_t)(i * 16), gbase, (uint32_t)(i * 16));
  }
  // Zero the x K-pad region (cols 0..31); only cols 0..2 rewritten per step.
  for (int i = tid; i < 16 * 32; i += 256)
    Ab[(i >> 5) * SA + (i & 31)] = f2bf(0.0f);
  // x(t=0) into A-buffer cols 0..2.
  if (tid < 48) {
    int r = tid / 3, c = tid - r * 3;
    Ab[r * SA + c] = f2bf(x[(size_t)(bt * 16 + r) * 3 + c]);
  }
  // Initial h into A-buffer cols 32..159.
  for (int i = tid; i < 16 * HS; i += 256) {
    int r = i >> 7, c = i & 127;
    Ab[r * SA + 32 + c] = f2bf(h0[(bt * 16 + r) * HS + c]);
  }
  // Initial c into accumulator-layout registers (persists all 100 steps).
  v8f zero = {};
  v8f creg = zero;
#pragma unroll
  for (int j = 0; j < 8; ++j)
    creg[j] = c0[(bt * 16 + j + mhi) * HS + 16 * w + N];

  // Per-lane biases for this wave's 16-wide slices.
  const float bI = bvec[        16 * w + N];
  const float bF = bvec[HS    + 16 * w + N];
  const float bG = bvec[2*HS  + 16 * w + N];
  const float bO = bvec[3*HS  + 16 * w + N];
  const float bP0 = bpc[(2 * w) * 16 + N];
  const float bP1 = bpc[(2 * w + 1) * 16 + N];

  const __bf16* wpcG = pk + PK_WPC;
  const __bf16* whdG = pk + PK_WHD;

  wait_asynccnt0();
  __syncthreads();

  for (int t = 0; t < T_STEPS; ++t) {
    // ---- Phase 2: gates[16,512] = [x|h] @ WU (wave w: cols 16w..16w+15 of
    // each of i/f/g/o).  Depth-1 software pipeline: K-tile kt+1 fragments load
    // while K-tile kt's 4 WMMAs execute.
    v8f acc[4] = {zero, zero, zero, zero};
    {
      v16bf a  = ldA(Ab, SA, lane, 0);
      v16bf b0 = ldB(wuL + ( 0 + w) * 512, lane);
      v16bf b1 = ldB(wuL + ( 8 + w) * 512, lane);
      v16bf b2 = ldB(wuL + (16 + w) * 512, lane);
      v16bf b3 = ldB(wuL + (24 + w) * 512, lane);
#pragma unroll
      for (int kt = 0; kt < 5; ++kt) {
        v16bf ac = a, c0 = b0, c1 = b1, c2 = b2, c3 = b3;
        if (kt + 1 < 5) {
          a = ldA(Ab, SA, lane, (kt + 1) * 32);
          const __bf16* tb = wuL + ((kt + 1) * 32 + w) * 512;
          b0 = ldB(tb,            lane);
          b1 = ldB(tb +  8 * 512, lane);
          b2 = ldB(tb + 16 * 512, lane);
          b3 = ldB(tb + 24 * 512, lane);
        }
        acc[0] = wmma_bf(ac, c0, acc[0]);
        acc[1] = wmma_bf(ac, c1, acc[1]);
        acc[2] = wmma_bf(ac, c2, acc[2]);
        acc[3] = wmma_bf(ac, c3, acc[3]);
      }
    }
    bar_signal();  // B2: this wave's reads of old h complete (consumed by WMMA)

    // Nonlinearities (pure VALU) hide barrier latency.
    float hv8[8];
#pragma unroll
    for (int j = 0; j < 8; ++j) {
      float iv = sigm(acc[0][j] + bI);
      float fv = sigm(acc[1][j] + bF);
      float gv = tanh_(acc[2][j] + bG);
      float ov = sigm(acc[3][j] + bO);
      float cv = fmaf(fv, creg[j], iv * gv);
      creg[j] = cv;
      hv8[j] = ov * tanh_(cv);
    }
    if (lane == 0 && w == 0 && t + 2 < T_STEPS) {
      __builtin_prefetch(&x[(size_t)((t + 2) * BATCH + bt * 16) * 3], 0, 1);
      __builtin_prefetch(&x[(size_t)((t + 2) * BATCH + bt * 16 + 10) * 3], 0, 1);
    }
    bar_wait();    // B2 complete: safe to overwrite h region

    // ---- Phase 3: publish new h (and next x) to LDS, stream h/c to HBM.
    size_t obase = (size_t)t * BATCH + bt * 16;
#pragma unroll
    for (int j = 0; j < 8; ++j)
      Ab[(j + mhi) * SA + 32 + 16 * w + N] = f2bf(hv8[j]);
    if (t + 1 < T_STEPS && tid < 48) {
      int r = tid / 3, c = tid - r * 3;
      Ab[r * SA + c] = f2bf(x[(size_t)((t + 1) * BATCH + bt * 16 + r) * 3 + c]);
    }
    wait_dscnt0();
    bar_signal();  // B3
#pragma unroll
    for (int j = 0; j < 8; ++j) {
      size_t go = (obase + j + mhi) * HS + 16 * w + N;
      __builtin_nontemporal_store(hv8[j], &oH[go]);
      __builtin_nontemporal_store(creg[j], &oC[go]);
    }
    bar_wait();    // B3 complete: h visible to all waves

    // ---- Phase 4: bn[16,256] = h @ Wbn.  Wave w owns N-tiles {2w, 2w+1}.
    v8f bacc0 = zero, bacc1 = zero;
    {
      v16bf a  = ldA(Ab, SA, lane, 32);
      v16bf b0 = ldB(wbnL + (2 * w)     * 512, lane);
      v16bf b1 = ldB(wbnL + (2 * w + 1) * 512, lane);
#pragma unroll
      for (int k = 0; k < 4; ++k) {
        v16bf ac = a, c0 = b0, c1 = b1;
        if (k + 1 < 4) {
          a  = ldA(Ab, SA, lane, 32 + (k + 1) * 32);
          b0 = ldB(wbnL + ((k + 1) * 16 + 2 * w)     * 512, lane);
          b1 = ldB(wbnL + ((k + 1) * 16 + 2 * w + 1) * 512, lane);
        }
        bacc0 = wmma_bf(ac, c0, bacc0);
        bacc1 = wmma_bf(ac, c1, bacc1);
      }
    }
#pragma unroll
    for (int j = 0; j < 8; ++j) {
      int M = j + mhi;
      BnB[M * SB + (2 * w) * 16 + N]      = f2bf(bacc0[j]);
      BnB[M * SB + (2 * w + 1) * 16 + N]  = f2bf(bacc1[j]);
    }
    wait_dscnt0();
    bar_signal();  // B4
#pragma unroll
    for (int j = 0; j < 8; ++j) {
      size_t go = (obase + j + mhi) * NBN;
      __builtin_nontemporal_store(bacc0[j], &oBn[go + (2 * w) * 16 + N]);
      __builtin_nontemporal_store(bacc1[j], &oBn[go + (2 * w + 1) * 16 + N]);
    }
    bar_wait();    // B4 complete: bn visible to all waves

    // ---- Phase 5: pc[16,256] = bn @ Wpc + b_pc ; hd = bn @ Whd + b_hd.
    // Depth-2 pipeline: Wpc tiles come from L2 (higher latency than LDS).
    v8f pacc0 = zero, pacc1 = zero;
    {
      v16bf pa[2], pb0[2], pb1[2];
      pa[0]  = ldA(BnB, SB, lane, 0);
      pb0[0] = ldB(wpcG + (2 * w)          * 512, lane);
      pb1[0] = ldB(wpcG + (2 * w + 1)      * 512, lane);
      pa[1]  = ldA(BnB, SB, lane, 32);
      pb0[1] = ldB(wpcG + (16 + 2 * w)     * 512, lane);
      pb1[1] = ldB(wpcG + (16 + 2 * w + 1) * 512, lane);
#pragma unroll
      for (int k = 0; k < 8; ++k) {
        v16bf ac = pa[k & 1], c0 = pb0[k & 1], c1 = pb1[k & 1];
        if (k + 2 < 8) {
          pa[k & 1]  = ldA(BnB, SB, lane, (k + 2) * 32);
          pb0[k & 1] = ldB(wpcG + ((k + 2) * 16 + 2 * w)     * 512, lane);
          pb1[k & 1] = ldB(wpcG + ((k + 2) * 16 + 2 * w + 1) * 512, lane);
        }
        pacc0 = wmma_bf(ac, c0, pacc0);
        pacc1 = wmma_bf(ac, c1, pacc1);
      }
    }
#pragma unroll
    for (int j = 0; j < 8; ++j) {
      size_t go = (obase + j + mhi) * NPC;
      __builtin_nontemporal_store(pacc0[j] + bP0, &oPc[go + (2 * w) * 16 + N]);
      __builtin_nontemporal_store(pacc1[j] + bP1, &oPc[go + (2 * w + 1) * 16 + N]);
    }
    if (w == 0) {  // wave-uniform branch: EXEC stays all-1s inside
      v8f hacc = zero;
      v16bf ha[2], hb[2];
      ha[0] = ldA(BnB, SB, lane, 0);
      hb[0] = ldB(whdG,       lane);
      ha[1] = ldA(BnB, SB, lane, 32);
      hb[1] = ldB(whdG + 512, lane);
#pragma unroll
      for (int k = 0; k < 8; ++k) {
        v16bf ac = ha[k & 1], c0 = hb[k & 1];
        if (k + 2 < 8) {
          ha[k & 1] = ldA(BnB, SB, lane, (k + 2) * 32);
          hb[k & 1] = ldB(whdG + (k + 2) * 512, lane);
        }
        hacc = wmma_bf(ac, c0, hacc);
      }
      if (N < NHD) {
#pragma unroll
        for (int j = 0; j < 8; ++j)
          __builtin_nontemporal_store(hacc[j] + bhd[N],
                                      &oHd[(obase + j + mhi) * NHD + N]);
      }
    }
    // No barrier here: Phase-5 reads BnB / next Phase-2 reads Ab only; the
    // next writes to Ab (h) and BnB are fenced by B2/B3 of the next step.
  }
}

// ---------------------------------------------------------------------------

extern "C" void kernel_launch(void* const* d_in, const int* in_sizes, int n_in,
                              void* d_out, int out_size, void* d_ws, size_t ws_size,
                              hipStream_t stream) {
  (void)in_sizes; (void)n_in; (void)out_size; (void)ws_size;
  const float* x    = (const float*)d_in[0];
  const float* ic   = (const float*)d_in[1];
  const float* W    = (const float*)d_in[2];
  const float* U    = (const float*)d_in[3];
  const float* b    = (const float*)d_in[4];
  const float* Wih  = (const float*)d_in[5];
  const float* bih  = (const float*)d_in[6];
  const float* Wic  = (const float*)d_in[7];
  const float* bic  = (const float*)d_in[8];
  const float* Wbn  = (const float*)d_in[9];
  const float* Wpc  = (const float*)d_in[10];
  const float* bpc  = (const float*)d_in[11];
  const float* Whd  = (const float*)d_in[12];
  const float* bhd  = (const float*)d_in[13];

  __bf16* pk = (__bf16*)d_ws;
  float* h0 = (float*)((char*)d_ws + WS_H0_BYTES);
  float* c0 = (float*)((char*)d_ws + WS_C0_BYTES);

  float* out = (float*)d_out;
  float* oHd = out;
  float* oPc = oHd + (size_t)T_STEPS * BATCH * NHD;
  float* oBn = oPc + (size_t)T_STEPS * BATCH * NPC;
  float* oH  = oBn + (size_t)T_STEPS * BATCH * NBN;
  float* oC  = oH  + (size_t)T_STEPS * BATCH * HS;

  k_pack<<<(PK_TOTAL + 255) / 256, 256, 0, stream>>>(W, U, Wbn, Wpc, Whd, pk);
  k_init<<<(BATCH * HS) / 256, 256, 0, stream>>>(ic, Wih, bih, Wic, bic, h0, c0);
  k_lstm<<<BATCH / 16, 256, L_TOTAL, stream>>>(x, b, bpc, bhd, pk, h0, c0,
                                               oHd, oPc, oBn, oH, oC);
}
